// GAT_TCN_85332410237515
// MI455X (gfx1250) — matrix operand order repaired and checked by reading, not compiled, not run on hardware
//
#include <hip/hip_runtime.h>
#include <hip/hip_bf16.h>

typedef __attribute__((ext_vector_type(16))) _Float16 v16h;
typedef __attribute__((ext_vector_type(8)))  _Float16 v8h;
typedef __attribute__((ext_vector_type(8)))  float    v8f;

#define NEG_SLOPE 0.2f

// ---------- helpers: monotonic float<->uint encoding for atomic segment-max ----------
__device__ __forceinline__ unsigned fenc(float f) {
    unsigned u = __float_as_uint(f);
    return (u & 0x80000000u) ? ~u : (u | 0x80000000u);
}
__device__ __forceinline__ float fdec(unsigned u) {
    unsigned b = (u & 0x80000000u) ? (u & 0x7FFFFFFFu) : ~u;
    return __uint_as_float(b);
}

// ---------- prep: pack w_src into WMMA B-fragment order; a_edge 3-vectors; zero pooled ----------
// packedB layout (per layer, 4096 halves): elem[((kb*4+ct)*32 + lane)*16 + j] =
//   w_src[K*64 + Ncol],  K = kb*32 + (lane>>4)*16 + j,  Ncol = ct*16 + (lane&15)
__global__ void prep_kernel(const float* __restrict__ ws0, const float* __restrict__ ws1,
                            const float* __restrict__ ws2,
                            const float* __restrict__ we0, const float* __restrict__ we1,
                            const float* __restrict__ we2,
                            const float* __restrict__ ae0, const float* __restrict__ ae1,
                            const float* __restrict__ ae2,
                            _Float16* __restrict__ pB, float* __restrict__ aev,
                            float* __restrict__ pooled) {
    int tid = threadIdx.x;
    const float* WS[3] = {ws0, ws1, ws2};
    for (int idx = tid; idx < 3 * 4096; idx += blockDim.x) {
        int li = idx / 4096, r = idx % 4096;
        int j = r & 15, L = (r >> 4) & 31, ct = (r >> 9) & 3, kb = r >> 11;
        int K = kb * 32 + ((L >> 4) << 4) + j;
        int Ncol = ct * 16 + (L & 15);
        pB[idx] = (_Float16)WS[li][K * 64 + Ncol];
    }
    const float* WE[3] = {we0, we1, we2};
    const float* AE[3] = {ae0, ae1, ae2};
    if (tid < 9) {
        int li = tid / 3, d = tid % 3;
        float s = 0.f;
        for (int c = 0; c < 64; c++) s += WE[li][d * 64 + c] * AE[li][c];
        aev[li * 4 + d] = s;
    }
    if (tid < 64) pooled[tid] = 0.f;
}

// ---------- node encoder: h = relu(x@W1+b1)@W2+b2 ; also h_half ----------
__global__ void encoder_kernel(const float* __restrict__ x,
                               const float* __restrict__ w1, const float* __restrict__ b1,
                               const float* __restrict__ w2, const float* __restrict__ b2,
                               float* __restrict__ h, _Float16* __restrict__ hh, int N) {
    __shared__ float sw1[160], sb1[32], sw2[2048], sb2[64];
    int tid = threadIdx.x;
    for (int i = tid; i < 160; i += blockDim.x) sw1[i] = w1[i];
    for (int i = tid; i < 32; i += blockDim.x) sb1[i] = b1[i];
    for (int i = tid; i < 2048; i += blockDim.x) sw2[i] = w2[i];
    for (int i = tid; i < 64; i += blockDim.x) sb2[i] = b2[i];
    __syncthreads();
    int n = blockIdx.x * blockDim.x + tid;
    if (n >= N) return;
    float xi[5];
    #pragma unroll
    for (int i = 0; i < 5; i++) xi[i] = x[(size_t)n * 5 + i];
    float t[32];
    #pragma unroll
    for (int o = 0; o < 32; o++) {
        float s = sb1[o];
        #pragma unroll
        for (int i = 0; i < 5; i++) s += xi[i] * sw1[i * 32 + o];
        t[o] = fmaxf(s, 0.f);
    }
    for (int o = 0; o < 64; o++) {
        float s = sb2[o];
        #pragma unroll
        for (int i = 0; i < 32; i++) s += t[i] * sw2[i * 64 + o];
        h[(size_t)n * 64 + o] = s;
        hh[(size_t)n * 64 + o] = (_Float16)s;
    }
}

// ---------- edge-weight MLP: 3 -> 32 relu -> 16 relu -> 1 sigmoid ----------
__global__ void edge_mlp_kernel(const float* __restrict__ ea,
                                const float* __restrict__ w1, const float* __restrict__ b1,
                                const float* __restrict__ w2, const float* __restrict__ b2,
                                const float* __restrict__ w3, const float* __restrict__ b3,
                                float* __restrict__ ew, int E) {
    __shared__ float sw1[96], sb1[32], sw2[512], sb2[16], sw3[16], sb3;
    int tid = threadIdx.x;
    for (int i = tid; i < 96; i += blockDim.x) sw1[i] = w1[i];
    for (int i = tid; i < 32; i += blockDim.x) sb1[i] = b1[i];
    for (int i = tid; i < 512; i += blockDim.x) sw2[i] = w2[i];
    for (int i = tid; i < 16; i += blockDim.x) sb2[i] = b2[i];
    for (int i = tid; i < 16; i += blockDim.x) sw3[i] = w3[i];
    if (tid == 0) sb3 = b3[0];
    __syncthreads();
    int e = blockIdx.x * blockDim.x + tid;
    if (e >= E) return;
    float a0 = ea[(size_t)e * 3], a1 = ea[(size_t)e * 3 + 1], a2 = ea[(size_t)e * 3 + 2];
    float t1[32];
    #pragma unroll
    for (int o = 0; o < 32; o++)
        t1[o] = fmaxf(sb1[o] + a0 * sw1[o] + a1 * sw1[32 + o] + a2 * sw1[64 + o], 0.f);
    float t2[16];
    #pragma unroll
    for (int o = 0; o < 16; o++) {
        float s = sb2[o];
        #pragma unroll
        for (int i = 0; i < 32; i++) s += t1[i] * sw2[i * 16 + o];
        t2[o] = fmaxf(s, 0.f);
    }
    float z = sb3;
    #pragma unroll
    for (int i = 0; i < 16; i++) z += t2[i] * sw3[i];
    ew[e] = 1.f / (1.f + __expf(-z));
}

// ---------- WMMA GEMM: xs = h_half (N x 64) @ w_src_half (64 x 64), fp32 accum ----------
// one wave -> one 16x16 D tile; K split into 2 steps of v_wmma_f32_16x16x32_f16.
// Requires EXEC all-ones: N must be a multiple of 32 (100000 is).
__global__ void gemm_xs_kernel(const _Float16* __restrict__ hh,
                               const _Float16* __restrict__ pB,
                               float* __restrict__ xs) {
    int tid = threadIdx.x;
    int wave = tid >> 5, lane = tid & 31;
    int ct = wave & 3;
    int rowchunk = blockIdx.x * 2 + (wave >> 2);
    int r0 = rowchunk * 16;
    int arow = r0 + (lane & 15);
    v8f c = {};
    #pragma unroll
    for (int kb = 0; kb < 2; kb++) {
        // A fragment: contiguous k = (lane>>4)*16 + j, same as packed B (lane group selects K halves)
        const _Float16* ap = hh + (size_t)arow * 64 + kb * 32 + ((lane >> 4) << 4);
        v8h lo = *(const v8h*)ap;
        v8h hi = *(const v8h*)(ap + 8);
        v16h a;
        #pragma unroll
        for (int j = 0; j < 8; j++) { a[j] = lo[j]; a[j + 8] = hi[j]; }
        // B fragment: pre-packed contiguous 32B per lane
        v16h b = *(const v16h*)(pB + ((size_t)((kb * 4 + ct) * 32 + lane)) * 16);
        c = __builtin_amdgcn_wmma_f32_16x16x32_f16(false, a, false, b, (short)0, c,
                                                   false, false);
    }
    // D: VGPR v -> row r0 + (lane>>4)*8 + v, col ct*16 + (lane&15)
    int col = ct * 16 + (lane & 15);
    int rbase = r0 + ((lane >> 4) << 3);
    #pragma unroll
    for (int v = 0; v < 8; v++) xs[(size_t)(rbase + v) * 64 + col] = c[v];
}

// ---------- per-node attention dots: a_src/a_dst = xs . att ----------
__global__ void dots_kernel(const float* __restrict__ xs,
                            const float* __restrict__ att_src,
                            const float* __restrict__ att_dst,
                            float* __restrict__ asrc, float* __restrict__ adst, int N) {
    int tid = threadIdx.x;
    int wave = tid >> 5, lane = tid & 31;
    int n = blockIdx.x * 8 + wave;
    if (n >= N) return;
    float x0 = xs[(size_t)n * 64 + lane];
    float x1 = xs[(size_t)n * 64 + lane + 32];
    float vs = x0 * att_src[lane] + x1 * att_src[lane + 32];
    float vd = x0 * att_dst[lane] + x1 * att_dst[lane + 32];
    #pragma unroll
    for (int off = 16; off > 0; off >>= 1) {
        vs += __shfl_down(vs, off, 32);
        vd += __shfl_down(vd, off, 32);
    }
    if (lane == 0) { asrc[n] = vs; adst[n] = vd; }
}

// ---------- per-layer init: zero accumulator, reset segmax/segsum ----------
__global__ void init_layer_kernel(float* __restrict__ outb, unsigned* __restrict__ mseg,
                                  float* __restrict__ ssum, int N) {
    size_t idx = (size_t)blockIdx.x * blockDim.x + threadIdx.x;
    size_t nf = (size_t)N * 64;
    if (idx < nf) outb[idx] = 0.f;
    if (idx < (size_t)N) { mseg[idx] = 0x007FFFFFu; /* enc(-inf) */ ssum[idx] = 0.f; }
}

// ---------- logits + segment max ----------
__global__ void logits_kernel(const int* __restrict__ ei, const float* __restrict__ ea,
                              const float* __restrict__ asrc, const float* __restrict__ adst,
                              const float* __restrict__ aev, float* __restrict__ lg,
                              unsigned* __restrict__ mseg, int E) {
    int e = blockIdx.x * blockDim.x + threadIdx.x;
    if (e >= E) return;
    int s = ei[e], d = ei[(size_t)E + e];
    float l = asrc[s] + adst[d]
            + ea[(size_t)e * 3] * aev[0] + ea[(size_t)e * 3 + 1] * aev[1]
            + ea[(size_t)e * 3 + 2] * aev[2];
    l = (l > 0.f) ? l : NEG_SLOPE * l;
    lg[e] = l;
    atomicMax(&mseg[d], fenc(l));
}

// ---------- exp + segment sum (also prefetch xs[src] rows for the scatter pass) ----------
__global__ void expsum_kernel(const int* __restrict__ ei, float* __restrict__ lg,
                              const unsigned* __restrict__ mseg, float* __restrict__ ssum,
                              const float* __restrict__ xs, int E) {
    int e = blockIdx.x * blockDim.x + threadIdx.x;
    if (e >= E) return;
    int s = ei[e], d = ei[(size_t)E + e];
    __builtin_prefetch(xs + (size_t)s * 64, 0, 1);  // warm L2 for aggregate_kernel
    float ex = __expf(lg[e] - fdec(mseg[d]));
    lg[e] = ex;
    atomicAdd(&ssum[d], ex);
}

// ---------- weighted scatter: out[dst] += alpha * xs[src], 64 lanes per edge ----------
__global__ void aggregate_kernel(const int* __restrict__ ei, const float* __restrict__ lg,
                                 const float* __restrict__ ssum, const float* __restrict__ ew,
                                 const float* __restrict__ xs, float* __restrict__ outb,
                                 int E) {
    int e = blockIdx.x * 4 + (threadIdx.x >> 6);
    int c = threadIdx.x & 63;
    if (e >= E) return;
    int s = ei[e], d = ei[(size_t)E + e];
    float alpha = lg[e] / (ssum[d] + 1e-16f) * ew[e];
    atomicAdd(outb + (size_t)d * 64 + c, xs[(size_t)s * 64 + c] * alpha);
}

// ---------- finalize layer: bias (+ELU for non-last), produce next h / h_half ----------
__global__ void finalize_kernel(const float* __restrict__ outb, const float* __restrict__ bias,
                                float* __restrict__ h, _Float16* __restrict__ hh,
                                int N, int is_last) {
    size_t idx = (size_t)blockIdx.x * blockDim.x + threadIdx.x;
    size_t nf = (size_t)N * 64;
    if (idx >= nf) return;
    int c = (int)(idx & 63);
    float v = outb[idx] + bias[c];
    if (!is_last) {
        v = (v > 0.f) ? v : (__expf(v) - 1.f);
        h[idx] = v;
        hh[idx] = (_Float16)v;
    } else {
        h[idx] = v;
    }
}

// ---------- pooled mean over nodes ----------
__global__ void pool_kernel(const float* __restrict__ h, float* __restrict__ pooled, int N) {
    __shared__ float buf[256];
    int tid = threadIdx.x;
    int c = tid & 63, g = tid >> 6;
    float s = 0.f;
    for (int n = blockIdx.x * 4 + g; n < N; n += gridDim.x * 4)
        s += h[(size_t)n * 64 + c];
    buf[tid] = s;
    __syncthreads();
    if (g == 0) {
        float t = buf[c] + buf[64 + c] + buf[128 + c] + buf[192 + c];
        atomicAdd(pooled + c, t);
    }
}

// ---------- TCN: two residual temporal blocks, single workgroup, all in LDS ----------
__global__ void tcn_kernel(const float* __restrict__ logf,
                           const float* __restrict__ w1a, const float* __restrict__ b1a,
                           const float* __restrict__ w2a, const float* __restrict__ b2a,
                           const float* __restrict__ wda, const float* __restrict__ bda,
                           const float* __restrict__ w1b, const float* __restrict__ b1b,
                           const float* __restrict__ w2b, const float* __restrict__ b2b,
                           const float* __restrict__ wdb, const float* __restrict__ bdb,
                           float* __restrict__ logfeat) {
    __shared__ float xin[64];
    __shared__ float A[64 * 64], B[64 * 64], Cb[64 * 64];
    int tid = threadIdx.x, nt = blockDim.x;
    for (int i = tid; i < 64; i += nt) xin[i] = logf[i];
    __syncthreads();
    // block 1 conv1 (cin=1, co=32) -> B
    for (int idx = tid; idx < 32 * 64; idx += nt) {
        int co = idx >> 6, l = idx & 63;
        float s = b1a[co];
        #pragma unroll
        for (int k = 0; k < 3; k++) {
            int p = l + k - 1;
            if (p >= 0 && p < 64) s += xin[p] * w1a[co * 3 + k];
        }
        B[idx] = fmaxf(s, 0.f);
    }
    __syncthreads();
    // block 1 conv2 (32,32,3) -> A
    for (int idx = tid; idx < 32 * 64; idx += nt) {
        int co = idx >> 6, l = idx & 63;
        float s = b2a[co];
        for (int ci = 0; ci < 32; ci++)
            #pragma unroll
            for (int k = 0; k < 3; k++) {
                int p = l + k - 1;
                if (p >= 0 && p < 64) s += B[ci * 64 + p] * w2a[(co * 32 + ci) * 3 + k];
            }
        A[idx] = fmaxf(s, 0.f);
    }
    __syncthreads();
    // block 1 residual + relu -> B (y1, 32x64)
    for (int idx = tid; idx < 32 * 64; idx += nt) {
        int co = idx >> 6, l = idx & 63;
        B[idx] = fmaxf(A[idx] + bda[co] + wda[co] * xin[l], 0.f);
    }
    __syncthreads();
    // block 2 conv1 (cin=32, co=64) -> A
    for (int idx = tid; idx < 64 * 64; idx += nt) {
        int co = idx >> 6, l = idx & 63;
        float s = b1b[co];
        for (int ci = 0; ci < 32; ci++)
            #pragma unroll
            for (int k = 0; k < 3; k++) {
                int p = l + k - 1;
                if (p >= 0 && p < 64) s += B[ci * 64 + p] * w1b[(co * 32 + ci) * 3 + k];
            }
        A[idx] = fmaxf(s, 0.f);
    }
    __syncthreads();
    // block 2 conv2 (64,64,3) -> Cb
    for (int idx = tid; idx < 64 * 64; idx += nt) {
        int co = idx >> 6, l = idx & 63;
        float s = b2b[co];
        for (int ci = 0; ci < 64; ci++)
            #pragma unroll
            for (int k = 0; k < 3; k++) {
                int p = l + k - 1;
                if (p >= 0 && p < 64) s += A[ci * 64 + p] * w2b[(co * 64 + ci) * 3 + k];
            }
        Cb[idx] = fmaxf(s, 0.f);
    }
    __syncthreads();
    // block 2 residual (wd: 64x32 1x1) + relu -> A (y2)
    for (int idx = tid; idx < 64 * 64; idx += nt) {
        int co = idx >> 6, l = idx & 63;
        float r = bdb[co];
        for (int ci = 0; ci < 32; ci++) r += B[ci * 64 + l] * wdb[co * 32 + ci];
        A[idx] = fmaxf(Cb[idx] + r, 0.f);
    }
    __syncthreads();
    // channel mean -> logfeat[l]
    if (tid < 64) {
        float s = 0.f;
        for (int co = 0; co < 64; co++) s += A[co * 64 + tid];
        logfeat[tid] = s * (1.f / 64.f);
    }
}

// ---------- fusion + LayerNorm + classifier + log_softmax (single workgroup) ----------
__global__ void fusion_kernel(const float* __restrict__ pooled, const float* __restrict__ logfeat,
                              const float* __restrict__ dur, const float* __restrict__ plen,
                              const float* __restrict__ W1, const float* __restrict__ B1,
                              const float* __restrict__ lng, const float* __restrict__ lnb,
                              const float* __restrict__ W2, const float* __restrict__ B2,
                              const float* __restrict__ CW1, const float* __restrict__ CB1,
                              const float* __restrict__ CW2, const float* __restrict__ CB2,
                              float* __restrict__ out, int N) {
    __shared__ float comb[130], z1[256], red[256], z2[128], z3[64], z4[10], stats[2];
    int tid = threadIdx.x;
    if (tid < 64) comb[tid] = pooled[tid] / (float)N;
    if (tid == 64) comb[64] = dur[0];
    if (tid == 65) comb[65] = plen[0];
    if (tid >= 66 && tid < 130) comb[tid] = logfeat[tid - 66];
    __syncthreads();
    float z = B1[tid];
    for (int i = 0; i < 130; i++) z += comb[i] * W1[i * 256 + tid];
    z = fmaxf(z, 0.f);
    z1[tid] = z;
    // mean
    red[tid] = z;
    __syncthreads();
    for (int s = 128; s > 0; s >>= 1) {
        if (tid < s) red[tid] += red[tid + s];
        __syncthreads();
    }
    if (tid == 0) stats[0] = red[0] / 256.f;
    __syncthreads();
    float mu = stats[0];
    float dz = z - mu;
    red[tid] = dz * dz;
    __syncthreads();
    for (int s = 128; s > 0; s >>= 1) {
        if (tid < s) red[tid] += red[tid + s];
        __syncthreads();
    }
    if (tid == 0) stats[1] = red[0] / 256.f;
    __syncthreads();
    float rstd = rsqrtf(stats[1] + 1e-5f);
    z1[tid] = dz * rstd * lng[tid] + lnb[tid];
    __syncthreads();
    if (tid < 128) {
        float s = B2[tid];
        for (int i = 0; i < 256; i++) s += z1[i] * W2[i * 128 + tid];
        z2[tid] = s;
    }
    __syncthreads();
    if (tid < 64) {
        float s = CB1[tid];
        for (int i = 0; i < 128; i++) s += z2[i] * CW1[i * 64 + tid];
        z3[tid] = fmaxf(s, 0.f);
    }
    __syncthreads();
    if (tid < 10) {
        float s = CB2[tid];
        for (int i = 0; i < 64; i++) s += z3[i] * CW2[i * 10 + tid];
        z4[tid] = s;
    }
    __syncthreads();
    if (tid == 0) {
        float mx = z4[0];
        for (int i = 1; i < 10; i++) mx = fmaxf(mx, z4[i]);
        float se = 0.f;
        for (int i = 0; i < 10; i++) se += __expf(z4[i] - mx);
        stats[0] = mx;
        stats[1] = __logf(se);
    }
    __syncthreads();
    if (tid < 10) out[tid] = z4[tid] - stats[0] - stats[1];
}

// =====================================================================================
extern "C" void kernel_launch(void* const* d_in, const int* in_sizes, int n_in,
                              void* d_out, int out_size, void* d_ws, size_t ws_size,
                              hipStream_t stream) {
    const float* x     = (const float*)d_in[0];
    const int*   ei    = (const int*)d_in[1];
    const float* eattr = (const float*)d_in[2];
    const float* logf  = (const float*)d_in[3];
    const float* dur   = (const float*)d_in[4];
    const float* plen  = (const float*)d_in[5];
    #define PRM(i) ((const float*)d_in[6 + (i)])
    // params flat order (insertion order of make_params):
    // 0..3 enc_w1,enc_b1,enc_w2,enc_b2 | 4..9 ew_w1..ew_b3
    // 10+6*li: w_src,att_src,att_dst,w_edge,att_edge,bias   (li=0..2)
    // 28+6*b:  w1,b1,w2,b2,wd,bd                            (b=0..1)
    // 40..49:  fus_w1,fus_b1,ln_g,ln_b,fus_w2,fus_b2,cls_w1,cls_b1,cls_w2,cls_b2

    const int N = in_sizes[0] / 5;
    const int E = in_sizes[1] / 2;
    const size_t nf = (size_t)N * 64;

    // workspace carve-up (floats)
    float* W = (float*)d_ws;
    float*    h    = W;               // nf
    float*    xs   = h + nf;          // nf
    float*    outb = xs + nf;         // nf
    float*    ew   = outb + nf;       // E
    float*    lg   = ew + E;          // E (logits -> exp, reused)
    float*    asrc = lg + E;          // N
    float*    adst = asrc + N;        // N
    unsigned* mseg = (unsigned*)(adst + N);   // N
    float*    ssum = (float*)(mseg + N);      // N
    _Float16* hh   = (_Float16*)(ssum + N);   // nf halves
    float*    tail = ssum + N + nf / 2;
    _Float16* pB   = (_Float16*)tail;         // 3*4096 halves
    float*    aev  = tail + 6144;             // 3*4
    float*    pooled  = aev + 16;             // 64
    float*    logfeat = pooled + 64;          // 64

    const int TB = 256;
    dim3 blk(TB);

    prep_kernel<<<1, blk, 0, stream>>>(PRM(10), PRM(16), PRM(22),
                                       PRM(13), PRM(19), PRM(25),
                                       PRM(14), PRM(20), PRM(26),
                                       pB, aev, pooled);
    encoder_kernel<<<(N + TB - 1) / TB, blk, 0, stream>>>(x, PRM(0), PRM(1), PRM(2), PRM(3),
                                                          h, hh, N);
    edge_mlp_kernel<<<(E + TB - 1) / TB, blk, 0, stream>>>(eattr, PRM(4), PRM(5), PRM(6),
                                                           PRM(7), PRM(8), PRM(9), ew, E);
    tcn_kernel<<<1, blk, 0, stream>>>(logf,
                                      PRM(28), PRM(29), PRM(30), PRM(31), PRM(32), PRM(33),
                                      PRM(34), PRM(35), PRM(36), PRM(37), PRM(38), PRM(39),
                                      logfeat);

    for (int li = 0; li < 3; li++) {
        int base = 10 + 6 * li;
        int is_last = (li == 2);
        gemm_xs_kernel<<<N / 32, blk, 0, stream>>>(hh, pB + (size_t)li * 4096, xs);
        dots_kernel<<<(N + 7) / 8, blk, 0, stream>>>(xs, PRM(base + 1), PRM(base + 2),
                                                     asrc, adst, N);
        init_layer_kernel<<<(int)((nf + TB - 1) / TB), blk, 0, stream>>>(outb, mseg, ssum, N);
        logits_kernel<<<(E + TB - 1) / TB, blk, 0, stream>>>(ei, eattr, asrc, adst,
                                                             aev + 4 * li, lg, mseg, E);
        expsum_kernel<<<(E + TB - 1) / TB, blk, 0, stream>>>(ei, lg, mseg, ssum, xs, E);
        aggregate_kernel<<<(E + 3) / 4, blk, 0, stream>>>(ei, lg, ssum, ew, xs, outb, E);
        finalize_kernel<<<(int)((nf + TB - 1) / TB), blk, 0, stream>>>(outb, PRM(base + 5),
                                                                       h, hh, N, is_last);
    }

    pool_kernel<<<128, blk, 0, stream>>>(h, pooled, N);
    fusion_kernel<<<1, blk, 0, stream>>>(pooled, logfeat, dur, plen,
                                         PRM(40), PRM(41), PRM(42), PRM(43),
                                         PRM(44), PRM(45), PRM(46), PRM(47),
                                         PRM(48), PRM(49),
                                         (float*)d_out, N);
    #undef PRM
}
